// TranssolverAttention_27333171872348
// MI455X (gfx1250) — compile-verified, matrix operand order checked by reading, not run on hardware
//
#include <hip/hip_runtime.h>
#include <hip/hip_bf16.h>
#include <stdint.h>

// ---------------------------------------------------------------------------
// Transsolver attention for MI455X (gfx1250).
// B=4, N=32768, D=256, H=8, S=64, DH=32.
// Strategy: bf16 WMMA (v_wmma_f32_16x16x32_bf16) for all large GEMMs,
// fp32 accumulation, fp32 output. Memory-bound problem -> bf16 intermediates.
// ---------------------------------------------------------------------------

#define B_   4
#define N_   32768
#define D_   256
#define H_   8
#define S_   64
#define DH_  32
#define HS_  512                  // H_*S_
#define ROWS_ (B_*N_)             // 131072
#define KCH_  64                  // split-K chunks in pooling
#define CHN_  (N_/KCH_)           // 512 n-elements per chunk
#define BH_   (B_*H_)             // 32

typedef __attribute__((ext_vector_type(16))) __bf16 v16bf;
typedef __attribute__((ext_vector_type(8)))  float  v8f;

// ---- workspace layout (bytes), all 256B aligned --------------------------
#define OFF_XBF     ((size_t)0)            // [ROWS, D] bf16            67,108,864
#define OFF_WFXT    ((size_t)67108864)     // [D, D]  bf16 (W_fx^T)        131,072
#define OFF_WCOMBT  ((size_t)67239936)     // [HS, D] bf16 (W_comb^T)      262,144
#define OFF_WOUTT   ((size_t)67502080)     // [D, D]  bf16 (W_out^T)       131,072
#define OFF_BCOMB   ((size_t)67633152)     // [HS] f32                       2,048
#define OFF_W       ((size_t)67635200)     // [ROWS, HS] bf16 slice w  134,217,728
#define OFF_WT      ((size_t)201852928)    // [B,H,S,N] bf16 w^T       134,217,728
#define OFF_FXT     ((size_t)336070656)    // [B,H,DH,N] bf16 fx^T      67,108,864
#define OFF_PART    ((size_t)403179520)    // [KCH,BH,S,DH] f32         16,777,216
#define OFF_NPART   ((size_t)419956736)    // [KCH,BH,S] f32               524,288
#define OFF_OSTT    ((size_t)420481024)    // [B,H,DH,S] bf16              131,072
// total ~420.6 MB

__device__ __forceinline__ unsigned short f2bf(float f) {
  unsigned u = __float_as_uint(f);
  u += 0x7FFFu + ((u >> 16) & 1u);          // round-to-nearest-even
  return (unsigned short)(u >> 16);
}
__device__ __forceinline__ float bf2f(unsigned short h) {
  return __uint_as_float(((unsigned)h) << 16);
}
__device__ __forceinline__ v8f wmma_bf16(v16bf a, v16bf b, v8f c) {
  return __builtin_amdgcn_wmma_f32_16x16x32_bf16(false, a, false, b,
                                                 (short)0, c, false, false);
}

// ---------------------------------------------------------------------------
// Prep kernels
// ---------------------------------------------------------------------------
__global__ void k_cvt_x(const float* __restrict__ x,
                        unsigned short* __restrict__ xbf, int n4) {
  for (int i = blockIdx.x * blockDim.x + threadIdx.x; i < n4;
       i += gridDim.x * blockDim.x) {
    float4 v = ((const float4*)x)[i];
    unsigned short tmp[4] __attribute__((aligned(8)));
    tmp[0] = f2bf(v.x); tmp[1] = f2bf(v.y);
    tmp[2] = f2bf(v.z); tmp[3] = f2bf(v.w);
    ((uint2*)xbf)[i] = *(const uint2*)tmp;
  }
}

// WT[n,k] = bf16(W[k,n]) for a 256x256 matrix
__global__ void k_transpose256(const float* __restrict__ W,
                               unsigned short* __restrict__ WT) {
  int id = blockIdx.x * blockDim.x + threadIdx.x;   // 65536 total
  int n = id >> 8, k = id & 255;
  WT[id] = f2bf(W[k * 256 + n]);
}

// W_comb^T[col,k] = (sum_c W_x[k, h*32+c] * W_slice[c, s]) / temp[h]
// b_comb[col]    = (sum_c b_x[h*32+c] * W_slice[c, s] + b_slice[s]) / temp[h]
__global__ void k_prep_comb(const float* __restrict__ W_x,
                            const float* __restrict__ b_x,
                            const float* __restrict__ W_slice,
                            const float* __restrict__ b_slice,
                            const float* __restrict__ temp,
                            unsigned short* __restrict__ WcombT,
                            float* __restrict__ bcomb) {
  int id = blockIdx.x * blockDim.x + threadIdx.x;   // 131072 total
  int col = id >> 8, k = id & 255;
  int h = col >> 6, s = col & 63;
  float invT = 1.0f / temp[h];
  float acc = 0.0f;
  for (int c = 0; c < DH_; ++c)
    acc += W_x[k * D_ + h * DH_ + c] * W_slice[c * S_ + s];
  WcombT[(size_t)col * D_ + k] = f2bf(acc * invT);
  if (id < HS_) {
    int h2 = id >> 6, s2 = id & 63;
    float bacc = b_slice[s2];
    for (int c = 0; c < DH_; ++c)
      bacc += b_x[h2 * DH_ + c] * W_slice[c * S_ + s2];
    bcomb[id] = bacc * (1.0f / temp[h2]);
  }
}

// ---------------------------------------------------------------------------
// Kernel 1: fused projections + softmax.
//   fx     = x @ W_fx + b_fx                       -> fxT  [B,H,DH,N] bf16
//   logits = x @ W_comb + b_comb (temp folded in)  -> softmax over S
//   w (slice_weights) stored both row-major [ROWS,HS] and transposed [B,H,S,N]
// 16 rows per workgroup, 4 waves; wave w owns fx cols [64w,64w+64) and
// logit cols [128w,128w+128).  K=256 -> 8 WMMA k-steps.
// ---------------------------------------------------------------------------
__global__ __launch_bounds__(128) void k_proj(
    const unsigned short* __restrict__ xbf,
    const unsigned short* __restrict__ WfxT,
    const unsigned short* __restrict__ WcombT,
    const float* __restrict__ b_fx,
    const float* __restrict__ bcomb,
    unsigned short* __restrict__ wsl,
    unsigned short* __restrict__ wslT,
    unsigned short* __restrict__ fxT) {
  __shared__ float lg[16][HS_];            // 32 KB
  __shared__ unsigned short fxl[16][D_];   //  8 KB
  const int t    = threadIdx.x;
  const int wv   = t >> 5;
  const int lane = t & 31;
  const int l15  = lane & 15;
  const int hoff = (lane >= 16) ? 16 : 0;
  const int rhalf = (lane >= 16) ? 8 : 0;
  const int row0 = blockIdx.x * 16;
  const int bb   = row0 >> 15;             // batch index (N = 2^15)
  const int n0   = row0 & (N_ - 1);

  v8f afx[4];
  v8f alg[8];
#pragma unroll
  for (int j = 0; j < 4; ++j) {
    float bias = b_fx[wv * 64 + j * 16 + l15];
#pragma unroll
    for (int e = 0; e < 8; ++e) afx[j][e] = bias;
  }
#pragma unroll
  for (int j = 0; j < 8; ++j) {
    float bias = bcomb[wv * 128 + j * 16 + l15];
#pragma unroll
    for (int e = 0; e < 8; ++e) alg[j][e] = bias;
  }

  const unsigned short* xrow = xbf + (size_t)(row0 + l15) * D_ + hoff;
#pragma unroll
  for (int k0 = 0; k0 < D_; k0 += 32) {
    v16bf a = *(const v16bf*)(xrow + k0);
#pragma unroll
    for (int j = 0; j < 4; ++j) {
      const unsigned short* bp =
          WfxT + (size_t)(wv * 64 + j * 16 + l15) * D_ + hoff + k0;
      afx[j] = wmma_bf16(a, *(const v16bf*)bp, afx[j]);
    }
#pragma unroll
    for (int j = 0; j < 8; ++j) {
      const unsigned short* bp =
          WcombT + (size_t)(wv * 128 + j * 16 + l15) * D_ + hoff + k0;
      alg[j] = wmma_bf16(a, *(const v16bf*)bp, alg[j]);
    }
  }

  // stage accumulators into LDS (C layout: VGPR e -> row e+rhalf, lane -> col)
#pragma unroll
  for (int j = 0; j < 4; ++j)
#pragma unroll
    for (int e = 0; e < 8; ++e)
      fxl[e + rhalf][wv * 64 + j * 16 + l15] = f2bf(afx[j][e]);
#pragma unroll
  for (int j = 0; j < 8; ++j)
#pragma unroll
    for (int e = 0; e < 8; ++e)
      lg[e + rhalf][wv * 128 + j * 16 + l15] = alg[j][e];
  __syncthreads();

  // softmax over S=64 per (row, head): 128 threads <-> 16 rows x 8 heads
  {
    const int m = t >> 3;
    const int h = t & 7;
    float* p = &lg[m][h * S_];
    float mx = -3.4e38f;
    for (int s = 0; s < S_; ++s) mx = fmaxf(mx, p[s]);
    float sum = 0.0f;
    for (int s = 0; s < S_; ++s) { float e = __expf(p[s] - mx); p[s] = e; sum += e; }
    float inv = 1.0f / sum;
    for (int s = 0; s < S_; ++s) p[s] *= inv;
  }
  __syncthreads();

  // slice weights, row-major bf16 [16,512]
  for (int cid = t; cid < 16 * 64; cid += 128) {
    int m = cid >> 6, seg = cid & 63;
    unsigned short tmp[8] __attribute__((aligned(16)));
#pragma unroll
    for (int e = 0; e < 8; ++e) tmp[e] = f2bf(lg[m][seg * 8 + e]);
    *(uint4*)(wsl + (size_t)(row0 + m) * HS_ + seg * 8) = *(const uint4*)tmp;
  }
  // transposed slice weights wT[b,h,s, n0..n0+15]
  for (int pid = t; pid < HS_; pid += 128) {
    int h = pid >> 6, s = pid & 63;
    unsigned short tmp[16] __attribute__((aligned(16)));
#pragma unroll
    for (int m = 0; m < 16; ++m) tmp[m] = f2bf(lg[m][pid]);
    unsigned short* dst = wslT + (((size_t)bb * H_ + h) * S_ + s) * N_ + n0;
    *(uint4*)(dst)     = *(const uint4*)(tmp);
    *(uint4*)(dst + 8) = *(const uint4*)(tmp + 8);
  }
  // transposed fx: fxT[b,h,c, n0..n0+15]
  for (int pid = t; pid < D_; pid += 128) {
    int h = pid >> 5, c = pid & 31;
    unsigned short tmp[16] __attribute__((aligned(16)));
#pragma unroll
    for (int m = 0; m < 16; ++m) tmp[m] = fxl[m][pid];
    unsigned short* dst = fxT + (((size_t)bb * H_ + h) * DH_ + c) * N_ + n0;
    *(uint4*)(dst)     = *(const uint4*)(tmp);
    *(uint4*)(dst + 8) = *(const uint4*)(tmp + 8);
  }
}

// ---------------------------------------------------------------------------
// Kernel 2: split-K pooling.  slice_token[s,c] = sum_n w[n,s]*fx[n,c]
// per (b,h): A = wT rows [S x Nchunk], B = fxT rows [DH x Nchunk].
// grid (KCH, BH), 4 waves; wave owns s-rows [16w,16w+16), 2 N'-tiles (DH=32).
// Also computes the per-chunk slice_norm partial.
// ---------------------------------------------------------------------------
__global__ __launch_bounds__(128) void k_pool(
    const unsigned short* __restrict__ wslT,
    const unsigned short* __restrict__ fxT,
    float* __restrict__ part,
    float* __restrict__ npart) {
  const int chunk = blockIdx.x;
  const int bh    = blockIdx.y;
  const int t    = threadIdx.x;
  const int wv   = t >> 5;
  const int lane = t & 31;
  const int l15  = lane & 15;
  const int hoff = (lane >= 16) ? 16 : 0;
  const int rhalf = (lane >= 16) ? 8 : 0;
  const int nbase = chunk * CHN_;
  const unsigned short* wTb = wslT + (size_t)bh * S_ * N_;
  const unsigned short* fxb = fxT + (size_t)bh * DH_ * N_;

  v8f acc0 = {};
  v8f acc1 = {};
  const unsigned short* arow = wTb + (size_t)(wv * 16 + l15) * N_ + nbase + hoff;
  const unsigned short* br0  = fxb + (size_t)(l15) * N_ + nbase + hoff;
  const unsigned short* br1  = fxb + (size_t)(16 + l15) * N_ + nbase + hoff;
#pragma unroll 4
  for (int k0 = 0; k0 < CHN_; k0 += 32) {
    v16bf a = *(const v16bf*)(arow + k0);
    acc0 = wmma_bf16(a, *(const v16bf*)(br0 + k0), acc0);
    acc1 = wmma_bf16(a, *(const v16bf*)(br1 + k0), acc1);
  }
  float* pb = part + ((size_t)chunk * BH_ + bh) * (S_ * DH_);
#pragma unroll
  for (int e = 0; e < 8; ++e) {
    int s = wv * 16 + e + rhalf;
    pb[s * DH_ + l15]       = acc0[e];
    pb[s * DH_ + 16 + l15]  = acc1[e];
  }
  // slice_norm partial: sum_n w[n,s] over this chunk
  __shared__ float nsum[2][S_];
  {
    int s = t & 63, hsel = t >> 6;
    const unsigned short* r = wTb + (size_t)s * N_ + nbase + hsel * (CHN_ / 2);
    float sum = 0.0f;
    for (int i = 0; i < CHN_ / 2; ++i) sum += bf2f(r[i]);
    nsum[hsel][s] = sum;
  }
  __syncthreads();
  if (t < S_) npart[((size_t)chunk * BH_ + bh) * S_ + t] = nsum[0][t] + nsum[1][t];
}

// ---------------------------------------------------------------------------
// Kernel 3: reduce split-K partials, normalize, tiny attention (S=64, DH=32).
// One workgroup per (b,h); everything fits in LDS; VALU fp32 (work is tiny).
// Stores out_slice_token transposed (ostT[b,h,c,s]) in bf16 for kernel 4's
// B operand.
// ---------------------------------------------------------------------------
__global__ __launch_bounds__(64) void k_attn(
    const float* __restrict__ part,
    const float* __restrict__ npart,
    const float* __restrict__ W_qkv,
    unsigned short* __restrict__ ostT) {
  const int bh = blockIdx.x;
  const int t  = threadIdx.x;
  __shared__ float st[S_][DH_];
  __shared__ float norm[S_];
  __shared__ float q[S_][DH_], kk[S_][DH_], vv[S_][DH_];
  __shared__ float at[S_][S_];
  float* stf = &st[0][0];

  for (int pid = t; pid < S_ * DH_; pid += 64) {
    float s = 0.0f;
    for (int ch = 0; ch < KCH_; ++ch)
      s += part[((size_t)ch * BH_ + bh) * (S_ * DH_) + pid];
    stf[pid] = s;
  }
  {
    float s = 0.0f;
    for (int ch = 0; ch < KCH_; ++ch)
      s += npart[((size_t)ch * BH_ + bh) * S_ + t];
    norm[t] = s;
  }
  __syncthreads();
  for (int pid = t; pid < S_ * DH_; pid += 64)
    stf[pid] /= (norm[pid >> 5] + 1e-5f);
  __syncthreads();

  {  // qkv projection, thread t = slice-token row
    int s = t;
    for (int c = 0; c < DH_; ++c) {
      float fq = 0.0f, fk = 0.0f, fv = 0.0f;
      for (int k = 0; k < DH_; ++k) {
        float x = st[s][k];
        fq += x * W_qkv[k * 96 + c];
        fk += x * W_qkv[k * 96 + 32 + c];
        fv += x * W_qkv[k * 96 + 64 + c];
      }
      q[s][c] = fq; kk[s][c] = fk; vv[s][c] = fv;
    }
  }
  __syncthreads();
  {  // attention row s = t
    int s = t;
    const float scale = 0.17677669529663687f;  // 32^-0.5
    float mx = -3.4e38f;
    for (int u = 0; u < S_; ++u) {
      float d = 0.0f;
      for (int k = 0; k < DH_; ++k) d += q[s][k] * kk[u][k];
      d *= scale;
      at[s][u] = d;
      mx = fmaxf(mx, d);
    }
    float sum = 0.0f;
    for (int u = 0; u < S_; ++u) { float e = __expf(at[s][u] - mx); at[s][u] = e; sum += e; }
    float inv = 1.0f / sum;
    for (int c = 0; c < DH_; ++c) {
      float o = 0.0f;
      for (int u = 0; u < S_; ++u) o += at[s][u] * vv[u][c];
      ostT[(size_t)bh * (DH_ * S_) + c * S_ + t] = f2bf(o * inv);
    }
  }
}

// ---------------------------------------------------------------------------
// Kernel 4: fused scatter + output projection.
//   out_x[n, h*32+c] = sum_s w[n, h*64+s] * ost[h][s,c]   (K=64, WMMA)
//   out = out_x @ W_out + b_out                            (K=256, WMMA)
// 16 rows per workgroup, 4 waves. Stage A: wave owns 2 heads. Stage B via LDS.
// ---------------------------------------------------------------------------
__global__ __launch_bounds__(128) void k_out(
    const unsigned short* __restrict__ wsl,
    const unsigned short* __restrict__ ostT,
    const unsigned short* __restrict__ WoutT,
    const float* __restrict__ b_out,
    float* __restrict__ out) {
  __shared__ unsigned short ox[16][D_];    // 8 KB, out_x tile in bf16
  const int t    = threadIdx.x;
  const int wv   = t >> 5;
  const int lane = t & 31;
  const int l15  = lane & 15;
  const int hoff = (lane >= 16) ? 16 : 0;
  const int rhalf = (lane >= 16) ? 8 : 0;
  const int row0 = blockIdx.x * 16;
  const int bb   = row0 >> 15;

  // ---- stage A: scatter back through slice weights -----------------------
#pragma unroll
  for (int hh = 0; hh < 2; ++hh) {
    int h = wv * 2 + hh;
    v8f acc0 = {};
    v8f acc1 = {};
    const unsigned short* arow = wsl + (size_t)(row0 + l15) * HS_ + h * S_ + hoff;
    const unsigned short* ob   = ostT + ((size_t)bb * H_ + h) * (DH_ * S_);
    const unsigned short* br0  = ob + (size_t)l15 * S_ + hoff;
    const unsigned short* br1  = ob + (size_t)(16 + l15) * S_ + hoff;
#pragma unroll
    for (int k0 = 0; k0 < S_; k0 += 32) {
      v16bf a = *(const v16bf*)(arow + k0);
      acc0 = wmma_bf16(a, *(const v16bf*)(br0 + k0), acc0);
      acc1 = wmma_bf16(a, *(const v16bf*)(br1 + k0), acc1);
    }
#pragma unroll
    for (int e = 0; e < 8; ++e) {
      int m = e + rhalf;
      ox[m][h * DH_ + l15]      = f2bf(acc0[e]);
      ox[m][h * DH_ + 16 + l15] = f2bf(acc1[e]);
    }
  }
  __syncthreads();

  // ---- stage B: final projection -----------------------------------------
  v8f acc[4];
#pragma unroll
  for (int j = 0; j < 4; ++j) {
    float bias = b_out[wv * 64 + j * 16 + l15];
#pragma unroll
    for (int e = 0; e < 8; ++e) acc[j][e] = bias;
  }
  const unsigned short* arow2 = &ox[l15][hoff];
#pragma unroll
  for (int k0 = 0; k0 < D_; k0 += 32) {
    v16bf a = *(const v16bf*)(arow2 + k0);
#pragma unroll
    for (int j = 0; j < 4; ++j) {
      const unsigned short* bp =
          WoutT + (size_t)(wv * 64 + j * 16 + l15) * D_ + hoff + k0;
      acc[j] = wmma_bf16(a, *(const v16bf*)bp, acc[j]);
    }
  }
#pragma unroll
  for (int j = 0; j < 4; ++j)
#pragma unroll
    for (int e = 0; e < 8; ++e)
      out[(size_t)(row0 + e + rhalf) * D_ + wv * 64 + j * 16 + l15] = acc[j][e];
}

// ---------------------------------------------------------------------------
extern "C" void kernel_launch(void* const* d_in, const int* in_sizes, int n_in,
                              void* d_out, int out_size, void* d_ws, size_t ws_size,
                              hipStream_t stream) {
  const float* x       = (const float*)d_in[0];
  const float* W_fx    = (const float*)d_in[1];
  const float* b_fx    = (const float*)d_in[2];
  const float* W_x     = (const float*)d_in[3];
  const float* b_x     = (const float*)d_in[4];
  const float* W_slice = (const float*)d_in[5];
  const float* b_slice = (const float*)d_in[6];
  const float* temp    = (const float*)d_in[7];
  const float* W_qkv   = (const float*)d_in[8];
  const float* W_out   = (const float*)d_in[9];
  const float* b_out   = (const float*)d_in[10];
  float* out = (float*)d_out;

  char* ws = (char*)d_ws;
  unsigned short* xbf    = (unsigned short*)(ws + OFF_XBF);
  unsigned short* WfxT   = (unsigned short*)(ws + OFF_WFXT);
  unsigned short* WcombT = (unsigned short*)(ws + OFF_WCOMBT);
  unsigned short* WoutT  = (unsigned short*)(ws + OFF_WOUTT);
  float*          bcomb  = (float*)(ws + OFF_BCOMB);
  unsigned short* wsl    = (unsigned short*)(ws + OFF_W);
  unsigned short* wslT   = (unsigned short*)(ws + OFF_WT);
  unsigned short* fxT    = (unsigned short*)(ws + OFF_FXT);
  float*          part   = (float*)(ws + OFF_PART);
  float*          npart  = (float*)(ws + OFF_NPART);
  unsigned short* ostT   = (unsigned short*)(ws + OFF_OSTT);

  // prep
  k_cvt_x<<<8192, 256, 0, stream>>>(x, xbf, ROWS_ * D_ / 4);
  k_transpose256<<<256, 256, 0, stream>>>(W_fx, WfxT);
  k_transpose256<<<256, 256, 0, stream>>>(W_out, WoutT);
  k_prep_comb<<<512, 256, 0, stream>>>(W_x, b_x, W_slice, b_slice, temp,
                                       WcombT, bcomb);
  // main pipeline
  k_proj<<<ROWS_ / 16, 128, 0, stream>>>(xbf, WfxT, WcombT, b_fx, bcomb,
                                         wsl, wslT, fxT);
  k_pool<<<dim3(KCH_, BH_), 128, 0, stream>>>(wslT, fxT, part, npart);
  k_attn<<<BH_, 64, 0, stream>>>(part, npart, W_qkv, ostT);
  k_out<<<ROWS_ / 16, 128, 0, stream>>>(wsl, ostT, WoutT, b_out, out);
}
